// ViewEncoder_214748364974
// MI455X (gfx1250) — matrix operand order repaired and checked by reading, not compiled, not run on hardware
//
#include <hip/hip_runtime.h>
#include <hip/hip_bf16.h>

// ---------------- problem constants ----------------
#define NN   50000      // nodes
#define EE   800000     // raw edges
#define ET   850000     // edges + self loops
#define INF_ 256
#define HH   4
#define CC   64
#define HC   256        // H*C
#define LAT  64
#define EPSL 1e-5f
#define SLOPE 0.2f

typedef __attribute__((ext_vector_type(16))) _Float16 v16h;
typedef __attribute__((ext_vector_type(8)))  _Float16 v8h;
typedef __attribute__((ext_vector_type(8)))  float    v8f;

// ==================================================================
// f32 -> f16 convert (activations / raw weights)
// ==================================================================
__global__ void k_f32_to_f16(const float* __restrict__ src, _Float16* __restrict__ dst, int n) {
    int i = blockIdx.x * 256 + threadIdx.x;
    if (i < n) dst[i] = (_Float16)src[i];
}

// ==================================================================
// Pack a KxNCOLS f32 weight into WMMA B-fragment layout (f16).
// Packed index: ((ct*8 + ks)*32 + lane)*16 + i
//   lane: n = lane&15, kblk = (lane>>4)*16
//   element i -> K = ks*32 + kblk + i, col = ct*16 + n
// (this code's k order, same as load_a_frag: lanes 0-15 K=0..15, lanes 16-31 K=16..31)
// ==================================================================
__global__ void k_pack_b(const float* __restrict__ W, _Float16* __restrict__ P, int ncols, int total) {
    int t = blockIdx.x * 256 + threadIdx.x;
    if (t >= total) return;
    int i    = t & 15;
    int lane = (t >> 4) & 31;
    int ks   = (t >> 9) & 7;
    int ct   = t >> 12;
    int n    = lane & 15;
    int kblk = (lane >> 4) * 16;
    int k    = ks * 32 + kblk + i;
    int col  = ct * 16 + n;
    P[t] = (_Float16)W[(size_t)k * ncols + col];
}

// ==================================================================
// WMMA fragment helpers
// ==================================================================
// A fragment (16x32 f16): lanes 0-15: M=lane, K = 0..15;
// lanes 16-31: M=lane-16, K = 16..31.
__device__ __forceinline__ v16h load_a_frag(const _Float16* __restrict__ A,
                                            int row0, int kbase, int lane, int ldk) {
    int m    = lane & 15;
    int koff = (lane >> 4) * 16;
    const _Float16* base = A + (size_t)(row0 + m) * ldk + kbase + koff;
    v8h lo = *(const v8h*)(base);
    v8h hi = *(const v8h*)(base + 8);
    v16h a;
#pragma unroll
    for (int i = 0; i < 8; ++i) { a[i] = lo[i]; a[i + 8] = hi[i]; }
    return a;
}

__device__ __forceinline__ v8f wmma_f16(v16h a, v16h b, v8f c) {
    return __builtin_amdgcn_wmma_f32_16x16x32_f16(false, a, false, b, (short)0, c, false, false);
}

// ==================================================================
// XL = A@Wl, XR = A@Wr   (A: NNxHC f16, W packed frags, out f32 NNxHC)
// block = 256 threads = 8 waves; wave -> one 16x16 tile; grid (3125, 2)
// ==================================================================
__global__ void k_gemm_xlxr(const _Float16* __restrict__ A,
                            const _Float16* __restrict__ BlP,
                            const _Float16* __restrict__ BrP,
                            float* __restrict__ XL, float* __restrict__ XR) {
    int lane = threadIdx.x & 31;
    int wave = threadIdx.x >> 5;
    int row0 = blockIdx.x * 16;
    int ct   = blockIdx.y * 8 + wave;      // column tile 0..15
    const v16h* Bl = (const v16h*)BlP;
    const v16h* Br = (const v16h*)BrP;
    v8f accl = {}; v8f accr = {};
#pragma unroll
    for (int ks = 0; ks < 8; ++ks) {
        v16h a  = load_a_frag(A, row0, ks * 32, lane, HC);
        v16h bl = Bl[(ct * 8 + ks) * 32 + lane];
        v16h br = Br[(ct * 8 + ks) * 32 + lane];
        accl = wmma_f16(a, bl, accl);
        accr = wmma_f16(a, br, accr);
    }
    int n  = lane & 15;
    int mb = (lane >> 4) * 8;
#pragma unroll
    for (int r = 0; r < 8; ++r) {
        size_t idx = (size_t)(row0 + mb + r) * HC + ct * 16 + n;
        XL[idx] = accl[r];
        XR[idx] = accr[r];
    }
}

// ==================================================================
// Edge score: score[e][h] = sum_c att[h,c]*leaky_relu(xl[src]+xr[dst])
// one wave per edge; lane -> 8 contiguous channels; head = lane>>3
// ==================================================================
__global__ void k_edge_score(const int* __restrict__ ei,
                             const float* __restrict__ XL, const float* __restrict__ XR,
                             const float* __restrict__ att, float* __restrict__ SC) {
    int lane = threadIdx.x & 31;
    int wave = threadIdx.x >> 5;
    int e = blockIdx.x * 8 + wave;                       // e < ET exactly
    int s = (e < EE) ? ei[e]      : (e - EE);
    int d = (e < EE) ? ei[EE + e] : (e - EE);
    int hc0 = lane * 8;
    const float* xl = XL + (size_t)s * HC + hc0;
    const float* xr = XR + (size_t)d * HC + hc0;
    const float* at = att + hc0;
    float p = 0.f;
#pragma unroll
    for (int i = 0; i < 8; ++i) {
        float v = xl[i] + xr[i];
        v = v > 0.f ? v : SLOPE * v;
        p += at[i] * v;
    }
    p += __shfl_xor(p, 1, 32);
    p += __shfl_xor(p, 2, 32);
    p += __shfl_xor(p, 4, 32);
    if ((lane & 7) == 0) SC[(size_t)e * HH + (lane >> 3)] = p;
}

// order-preserving float<->uint mapping for atomic max
__device__ __forceinline__ unsigned fmap(float f) {
    unsigned u = __float_as_uint(f);
    return (u & 0x80000000u) ? ~u : (u | 0x80000000u);
}
__device__ __forceinline__ float funmap(unsigned k) {
    unsigned u = (k & 0x80000000u) ? (k ^ 0x80000000u) : ~k;
    return __uint_as_float(u);
}

// segment max over dst (mapped-uint atomicMax; M pre-zeroed, self-loops
// guarantee every segment is written)
__global__ void k_seg_max(const int* __restrict__ ei, const float* __restrict__ SC,
                          unsigned* __restrict__ M) {
    int idx = blockIdx.x * 256 + threadIdx.x;
    if (idx >= ET * HH) return;
    int e = idx >> 2, h = idx & 3;
    int d = (e < EE) ? ei[EE + e] : (e - EE);
    atomicMax(&M[(size_t)d * HH + h], fmap(SC[idx]));
}

// ex = exp(score - m[dst]); denom[dst] += ex (overwrite SC with ex)
__global__ void k_exp_den(const int* __restrict__ ei, float* __restrict__ SC,
                          const unsigned* __restrict__ M, float* __restrict__ DEN) {
    int idx = blockIdx.x * 256 + threadIdx.x;
    if (idx >= ET * HH) return;
    int e = idx >> 2, h = idx & 3;
    int d = (e < EE) ? ei[EE + e] : (e - EE);
    float ex = __expf(SC[idx] - funmap(M[(size_t)d * HH + h]));
    SC[idx] = ex;
    atomicAdd(&DEN[(size_t)d * HH + h], ex);
}

// AGG[dst] += (ex/denom[dst]) * xl[src]; one wave per edge
__global__ void k_aggregate(const int* __restrict__ ei, const float* __restrict__ EX,
                            const float* __restrict__ DEN, const float* __restrict__ XL,
                            float* __restrict__ AGG) {
    int lane = threadIdx.x & 31;
    int wave = threadIdx.x >> 5;
    int e = blockIdx.x * 8 + wave;
    int s = (e < EE) ? ei[e]      : (e - EE);
    int d = (e < EE) ? ei[EE + e] : (e - EE);
    int h = lane >> 3;
    float alpha = EX[(size_t)e * HH + h] / DEN[(size_t)d * HH + h];
    int hc0 = lane * 8;
    const float* xl = XL + (size_t)s * HC + hc0;
    float* out = AGG + (size_t)d * HC + hc0;
#pragma unroll
    for (int i = 0; i < 8; ++i) atomicAdd(&out[i], alpha * xl[i]);
}

// ==================================================================
// h = elu(layernorm(AGG + bias)) -> f16 (input of next GEMM)
// one wave per node; 8 vals/lane; full-wave shuffle reduction
// ==================================================================
__global__ void k_bias_ln_elu(const float* __restrict__ AGG, const float* __restrict__ bias,
                              const float* __restrict__ g, const float* __restrict__ b,
                              _Float16* __restrict__ out) {
    int lane = threadIdx.x & 31;
    int wave = threadIdx.x >> 5;
    int nrow = blockIdx.x * 8 + wave;
    int hc0  = lane * 8;
    const float* row = AGG + (size_t)nrow * HC;
    float y[8]; float s = 0.f, sq = 0.f;
#pragma unroll
    for (int i = 0; i < 8; ++i) {
        float v = row[hc0 + i] + bias[hc0 + i];
        y[i] = v; s += v; sq += v * v;
    }
#pragma unroll
    for (int o = 1; o < 32; o <<= 1) {
        s  += __shfl_xor(s,  o, 32);
        sq += __shfl_xor(sq, o, 32);
    }
    float mu  = s  * (1.f / HC);
    float var = sq * (1.f / HC) - mu * mu;
    float r   = rsqrtf(var + EPSL);
    _Float16* orow = out + (size_t)nrow * HC;
#pragma unroll
    for (int i = 0; i < 8; ++i) {
        float z = (y[i] - mu) * r * g[hc0 + i] + b[hc0 + i];
        z = z > 0.f ? z : (__expf(z) - 1.f);
        orow[hc0 + i] = (_Float16)z;
    }
}

// ==================================================================
// mu = h@Wmu + bmu ; logvar = h@Wlv + blv  -> d_out (mu | logvar)
// 8 waves: waves 0-3 -> mu col tiles, 4-7 -> logvar col tiles
// ==================================================================
__global__ void k_final_gemm(const _Float16* __restrict__ A,
                             const _Float16* __restrict__ BmuP, const _Float16* __restrict__ BlvP,
                             const float* __restrict__ bmu, const float* __restrict__ blv,
                             float* __restrict__ out) {
    int lane  = threadIdx.x & 31;
    int wave  = threadIdx.x >> 5;
    int row0  = blockIdx.x * 16;
    int which = wave >> 2;            // 0 = mu, 1 = logvar
    int ct    = wave & 3;             // column tile (LAT/16 = 4)
    const v16h* B = (const v16h*)(which ? BlvP : BmuP);
    const float* bias = which ? blv : bmu;
    float* dst = out + (size_t)which * (size_t)NN * LAT;
    v8f acc = {};
#pragma unroll
    for (int ks = 0; ks < 8; ++ks) {
        v16h a = load_a_frag(A, row0, ks * 32, lane, HC);
        acc = wmma_f16(a, B[(ct * 8 + ks) * 32 + lane], acc);
    }
    int n  = lane & 15;
    int mb = (lane >> 4) * 8;
    float bv = bias[ct * 16 + n];
#pragma unroll
    for (int r = 0; r < 8; ++r)
        dst[(size_t)(row0 + mb + r) * LAT + ct * 16 + n] = acc[r] + bv;
}

// ==================================================================
// host launcher
// ==================================================================
extern "C" void kernel_launch(void* const* d_in, const int* in_sizes, int n_in,
                              void* d_out, int out_size, void* d_ws, size_t ws_size,
                              hipStream_t stream) {
    (void)in_sizes; (void)n_in; (void)out_size; (void)ws_size;
    const float* x    = (const float*)d_in[0];
    const int*   ei   = (const int*)  d_in[1];
    const float* Wl[2]  = { (const float*)d_in[2],  (const float*)d_in[8]  };
    const float* Wr[2]  = { (const float*)d_in[3],  (const float*)d_in[9]  };
    const float* att[2] = { (const float*)d_in[4],  (const float*)d_in[10] };
    const float* bia[2] = { (const float*)d_in[5],  (const float*)d_in[11] };
    const float* lng[2] = { (const float*)d_in[6],  (const float*)d_in[12] };
    const float* lnb[2] = { (const float*)d_in[7],  (const float*)d_in[13] };
    const float* Wmu = (const float*)d_in[14];
    const float* bmu = (const float*)d_in[15];
    const float* Wlv = (const float*)d_in[16];
    const float* blv = (const float*)d_in[17];
    float* out = (float*)d_out;

    // ---- workspace layout (all 256B-aligned sizes) ----
    char* ws = (char*)d_ws;
    size_t off = 0;
    _Float16* Ah   = (_Float16*)(ws + off); off += (size_t)NN * HC * 2;      // 25.6 MB
    _Float16* W1h  = (_Float16*)(ws + off); off += (size_t)HC * HC * 2;      // packed Wl
    _Float16* W2h  = (_Float16*)(ws + off); off += (size_t)HC * HC * 2;      // packed Wr
    _Float16* Wmuh = (_Float16*)(ws + off); off += (size_t)HC * LAT * 2;
    _Float16* Wlvh = (_Float16*)(ws + off); off += (size_t)HC * LAT * 2;
    float*    XL   = (float*)(ws + off);    off += (size_t)NN * HC * 4;      // 51.2 MB
    float*    XR   = (float*)(ws + off);    off += (size_t)NN * HC * 4;
    float*    AGG  = (float*)(ws + off);    off += (size_t)NN * HC * 4;
    float*    SC   = (float*)(ws + off);    off += (size_t)ET * HH * 4;      // 13.6 MB
    unsigned* M    = (unsigned*)(ws + off); off += (size_t)NN * HH * 4;
    float*    DEN  = (float*)(ws + off);    off += (size_t)NN * HH * 4;

    const int nhc = NN * HC;                       // 12,800,000
    k_f32_to_f16<<<(nhc + 255) / 256, 256, 0, stream>>>(x, Ah, nhc);

    for (int L = 0; L < 2; ++L) {
        k_pack_b<<<(HC * HC + 255) / 256, 256, 0, stream>>>(Wl[L], W1h, HC, HC * HC);
        k_pack_b<<<(HC * HC + 255) / 256, 256, 0, stream>>>(Wr[L], W2h, HC, HC * HC);
        k_gemm_xlxr<<<dim3(NN / 16, 2), 256, 0, stream>>>(Ah, W1h, W2h, XL, XR);

        hipMemsetAsync(M,   0, (size_t)NN * HH * 4, stream);
        hipMemsetAsync(DEN, 0, (size_t)NN * HH * 4, stream);
        hipMemsetAsync(AGG, 0, (size_t)NN * HC * 4, stream);

        k_edge_score<<<ET / 8, 256, 0, stream>>>(ei, XL, XR, att[L], SC);
        k_seg_max  <<<(ET * HH + 255) / 256, 256, 0, stream>>>(ei, SC, M);
        k_exp_den  <<<(ET * HH + 255) / 256, 256, 0, stream>>>(ei, SC, M, DEN);
        k_aggregate<<<ET / 8, 256, 0, stream>>>(ei, SC, DEN, XL, AGG);
        k_bias_ln_elu<<<NN / 8, 256, 0, stream>>>(AGG, bia[L], lng[L], lnb[L], Ah);
    }

    k_pack_b<<<(HC * LAT + 255) / 256, 256, 0, stream>>>(Wmu, Wmuh, LAT, HC * LAT);
    k_pack_b<<<(HC * LAT + 255) / 256, 256, 0, stream>>>(Wlv, Wlvh, LAT, HC * LAT);
    k_final_gemm<<<NN / 16, 256, 0, stream>>>(Ah, Wmuh, Wlvh, bmu, blv, out);
}
